// GroupedQueryAttention_13855564497552
// MI455X (gfx1250) — matrix operand order repaired and checked by reading, not compiled, not run on hardware
//
#include <hip/hip_runtime.h>
#include <hip/hip_bf16.h>

// ---------------- problem constants ----------------
#define B_   2
#define S_   2048
#define HID_ 1536
#define NH_  24
#define NKV_ 6
#define HD_  64
#define G_   (NH_ / NKV_)     // 4
#define NTOK (B_ * S_)        // 4096

typedef __attribute__((ext_vector_type(16))) _Float16 v16h;
typedef __attribute__((ext_vector_type(8)))  _Float16 v8h;
typedef __attribute__((ext_vector_type(8)))  float    v8f;

union F16x16 { v16h v; v8h h[2]; };

__device__ __forceinline__ v8f wmma_f16(v16h a, v16h b, v8f c) {
    // D = A(16x32 f16) * B(32x16 f16) + C(16x16 f32)
    return __builtin_amdgcn_wmma_f32_16x16x32_f16(
        /*neg_a=*/false, a, /*neg_b=*/false, b,
        /*c_mod=*/(short)0, c, /*reuse_a=*/false, /*reuse_b=*/false);
}

// ---------------- 0) fp32 -> f16 pack ----------------
__global__ void pack_f16_kernel(const float* __restrict__ in,
                                _Float16* __restrict__ out, int n) {
    int i = blockIdx.x * blockDim.x + threadIdx.x;
    int stride = gridDim.x * blockDim.x;
    for (; i < n; i += stride) out[i] = (_Float16)in[i];
}

// ---------------- 1) fused QKV projection + RoPE ----------------
// grid = (NTOK/32, NH+2*NKV), block = 32 (one wave)
// Each wave: 32 tokens x one 64-wide head column; K = HID in 32-steps,
// software-pipelined (ping-pong fragment buffers).
__global__ __launch_bounds__(32) void qkv_rope_kernel(
    const _Float16* __restrict__ Xh,
    const _Float16* __restrict__ Wqh,
    const _Float16* __restrict__ Wkh,
    const _Float16* __restrict__ Wvh,
    const int*      __restrict__ pos_ids,
    _Float16* __restrict__ Qh,   // [B, NH, S, HD]
    _Float16* __restrict__ Kh,   // [B, NKV, S, HD]
    _Float16* __restrict__ Vt)   // [B, NKV, HD, S]  (transposed)
{
    const int lane  = threadIdx.x & 31;
    const int n     = lane & 15;
    const int hi    = lane >> 4;
    const int koffA = hi ? 16 : 0;
    const int koffB = hi ? 16 : 0;

    const int mt = blockIdx.x;            // 32-token row tile
    const int jh = blockIdx.y;            // 0..35: Q heads, K heads, V heads

    const _Float16* W;
    int mode, head;
    if (jh < NH_)             { W = Wqh + (size_t)jh * HD_ * HID_;                mode = 0; head = jh; }
    else if (jh < NH_ + NKV_) { W = Wkh + (size_t)(jh - NH_) * HD_ * HID_;        mode = 1; head = jh - NH_; }
    else                      { W = Wvh + (size_t)(jh - NH_ - NKV_) * HD_ * HID_; mode = 2; head = jh - NH_ - NKV_; }

    const _Float16* arow[2] = {
        Xh + (size_t)(mt * 32 +      n) * HID_,
        Xh + (size_t)(mt * 32 + 16 + n) * HID_
    };

    v8f acc[2][4] = {};
    F16x16 aP[2], bP[4], aQ[2], bQ[4];

    auto loadAB = [&](F16x16 (&aa)[2], F16x16 (&bb)[4], int k0) {
#pragma unroll
        for (int mi = 0; mi < 2; ++mi) {
            aa[mi].h[0] = *(const v8h*)(arow[mi] + k0 + koffA);
            aa[mi].h[1] = *(const v8h*)(arow[mi] + k0 + 8 + koffA);
        }
#pragma unroll
        for (int jj = 0; jj < 4; ++jj) {
            const _Float16* brow = W + (size_t)(jj * 16 + n) * HID_ + k0 + koffB;
            bb[jj].h[0] = *(const v8h*)(brow);
            bb[jj].h[1] = *(const v8h*)(brow + 8);
        }
    };
    auto mma8 = [&](F16x16 (&aa)[2], F16x16 (&bb)[4]) {
#pragma unroll
        for (int jj = 0; jj < 4; ++jj) {
            acc[0][jj] = wmma_f16(aa[0].v, bb[jj].v, acc[0][jj]);
            acc[1][jj] = wmma_f16(aa[1].v, bb[jj].v, acc[1][jj]);
        }
    };

    loadAB(aP, bP, 0);
    for (int k0 = 0; k0 < HID_; k0 += 64) {       // two 32-steps per iter
        loadAB(aQ, bQ, k0 + 32);
        mma8(aP, bP);
        if (k0 + 64 < HID_) loadAB(aP, bP, k0 + 64);
        mma8(aQ, bQ);
    }

    if (mode == 2) {
        // V: store transposed [b, kv, d, s]
#pragma unroll
        for (int mi = 0; mi < 2; ++mi)
#pragma unroll
            for (int jj = 0; jj < 4; ++jj) {
                int d = jj * 16 + n;
#pragma unroll
                for (int r = 0; r < 8; ++r) {
                    int token = mt * 32 + mi * 16 + r + hi * 8;
                    int b = token >> 11;           // /S_
                    int s = token & (S_ - 1);
                    Vt[((size_t)(b * NKV_ + head) * HD_ + d) * S_ + s] =
                        (_Float16)acc[mi][jj][r];
                }
            }
    } else {
        // Q/K: RoPE in registers (d and d+32 live in acc[.][jj] and acc[.][jj+2])
        const float klog = -logf(10000.0f) / 32.0f;   // theta = pos * exp(d * klog)
#pragma unroll
        for (int mi = 0; mi < 2; ++mi)
#pragma unroll
            for (int r = 0; r < 8; ++r) {
                int token = mt * 32 + mi * 16 + r + hi * 8;
                int b = token >> 11;
                int s = token & (S_ - 1);
                float pos = (float)pos_ids[b * S_ + s];
                _Float16* dst = (mode == 0)
                    ? Qh + ((size_t)(b * NH_  + head) * S_ + s) * HD_
                    : Kh + ((size_t)(b * NKV_ + head) * S_ + s) * HD_;
#pragma unroll
                for (int jj = 0; jj < 2; ++jj) {
                    int d = jj * 16 + n;                       // 0..31
                    float th = pos * __expf((float)d * klog);
                    float c  = __cosf(th), sn = __sinf(th);
                    float x1 = acc[mi][jj][r], x2 = acc[mi][jj + 2][r];
                    dst[d]      = (_Float16)(x1 * c - x2 * sn);
                    dst[d + 32] = (_Float16)(x2 * c + x1 * sn);
                }
            }
    }
}

// ---------------- 2) causal flash attention ----------------
// grid = (S/64, NH, B), block = 128 (4 waves, each wave owns 16 query rows)
__global__ __launch_bounds__(128) void attn_kernel(
    const _Float16* __restrict__ Qh,   // [B, NH, S, HD]
    const _Float16* __restrict__ Kh,   // [B, NKV, S, HD]
    const _Float16* __restrict__ Vt,   // [B, NKV, HD, S]
    _Float16* __restrict__ AttnOut)    // [B*S, NH*HD]
{
    __shared__ __align__(16) _Float16 pl[4][16 * 32];

    const int lane  = threadIdx.x & 31;
    const int wave  = threadIdx.x >> 5;
    const int n     = lane & 15;
    const int hi    = lane >> 4;
    const int koffA = hi ? 16 : 0;
    const int koffB = hi ? 16 : 0;

    const int h  = blockIdx.y;
    const int b  = blockIdx.z;
    const int kv = h / G_;
    const int q0 = blockIdx.x * 64 + wave * 16;

    const _Float16* Qb = Qh + ((size_t)(b * NH_  + h ) * S_) * HD_;
    const _Float16* Kb = Kh + ((size_t)(b * NKV_ + kv) * S_) * HD_;
    const _Float16* Vb = Vt + ((size_t)(b * NKV_ + kv) * HD_) * S_;

    // Preload Q fragments for the whole row-tile (16 x 64 = 2 k-steps)
    F16x16 qf[2];
    {
        const _Float16* qrow = Qb + (size_t)(q0 + n) * HD_;
#pragma unroll
        for (int t = 0; t < 2; ++t) {
            qf[t].h[0] = *(const v8h*)(qrow + t * 32 + koffA);
            qf[t].h[1] = *(const v8h*)(qrow + t * 32 + 8 + koffA);
        }
    }

    float rowM[8], rowS[8];
    v8f oacc[4] = {};
#pragma unroll
    for (int r = 0; r < 8; ++r) { rowM[r] = -3.0e38f; rowS[r] = 0.0f; }
    const float scale = 0.125f;    // 1/sqrt(64)

    _Float16* pw = &pl[wave][0];

    for (int kb = 0; kb <= q0 + 15; kb += 32) {
        // --- issue ALL global loads for this key block up front:           ---
        // --- 4 K fragments + 4 V fragments; V stays in flight over softmax ---
        F16x16 kf[2][2], vf[4];
#pragma unroll
        for (int t = 0; t < 2; ++t) {
            const _Float16* kr0 = Kb + (size_t)(kb + n)      * HD_ + t * 32 + koffB;
            const _Float16* kr1 = Kb + (size_t)(kb + 16 + n) * HD_ + t * 32 + koffB;
            kf[t][0].h[0] = *(const v8h*)(kr0); kf[t][0].h[1] = *(const v8h*)(kr0 + 8);
            kf[t][1].h[0] = *(const v8h*)(kr1); kf[t][1].h[1] = *(const v8h*)(kr1 + 8);
        }
#pragma unroll
        for (int jj = 0; jj < 4; ++jj) {
            const _Float16* vr = Vb + (size_t)(jj * 16 + n) * S_ + kb + koffB;
            vf[jj].h[0] = *(const v8h*)(vr);
            vf[jj].h[1] = *(const v8h*)(vr + 8);
        }

        // --- scores: two 16x16 tiles covering keys [kb, kb+32) ---
        v8f sc0 = {}, sc1 = {};
#pragma unroll
        for (int t = 0; t < 2; ++t) {
            sc0 = wmma_f16(qf[t].v, kf[t][0].v, sc0);
            sc1 = wmma_f16(qf[t].v, kf[t][1].v, sc1);
        }

        // --- online softmax (row = 16 lanes of one half-wave, 8 rows per lane) ---
        float p0[8], p1[8];
#pragma unroll
        for (int r = 0; r < 8; ++r) {
            int qg = q0 + r + hi * 8;
            float s0 = sc0[r] * scale; if (kb + n      > qg) s0 = -3.0e38f;
            float s1 = sc1[r] * scale; if (kb + 16 + n > qg) s1 = -3.0e38f;
            float mx = fmaxf(s0, s1);
#pragma unroll
            for (int msk = 1; msk < 16; msk <<= 1)
                mx = fmaxf(mx, __shfl_xor(mx, msk, 32));
            float newM = fmaxf(rowM[r], mx);
            float corr = __expf(rowM[r] - newM);
            rowM[r] = newM;
            p0[r] = __expf(s0 - newM);
            p1[r] = __expf(s1 - newM);
            float ps = p0[r] + p1[r];
#pragma unroll
            for (int msk = 1; msk < 16; msk <<= 1)
                ps += __shfl_xor(ps, msk, 32);
            rowS[r] = rowS[r] * corr + ps;
#pragma unroll
            for (int jj = 0; jj < 4; ++jj) oacc[jj][r] *= corr;
        }

        // --- relayout P (C-layout f32 -> A-layout f16) via per-wave LDS ---
#pragma unroll
        for (int r = 0; r < 8; ++r) {
            int m = r + hi * 8;
            pw[m * 32 + n]      = (_Float16)p0[r];
            pw[m * 32 + 16 + n] = (_Float16)p1[r];
        }
        // same-wave DS ops are in-order; read back as A fragment (m = n row)
        F16x16 pa;
        pa.h[0] = *(const v8h*)(pw + n * 32 + koffA);
        pa.h[1] = *(const v8h*)(pw + n * 32 + 8 + koffA);

        // --- P @ V : fragments already in registers ---
#pragma unroll
        for (int jj = 0; jj < 4; ++jj)
            oacc[jj] = wmma_f16(pa.v, vf[jj].v, oacc[jj]);
    }

    // --- normalize + store (f16, row-major [token, NH*HD]) ---
#pragma unroll
    for (int r = 0; r < 8; ++r) {
        float inv = 1.0f / rowS[r];
        int sq = q0 + r + hi * 8;
        _Float16* orow = AttnOut + ((size_t)b * S_ + sq) * (NH_ * HD_) + h * HD_;
#pragma unroll
        for (int jj = 0; jj < 4; ++jj)
            orow[jj * 16 + n] = (_Float16)(oacc[jj][r] * inv);
    }
}

// ---------------- 3) output projection ----------------
// grid = (NTOK/32, HID/64), block = 32. out = AttnOut @ Wo^T (fp32 out),
// same pipelined 32x64 wave tile as the QKV GEMM.
__global__ __launch_bounds__(32) void oproj_kernel(
    const _Float16* __restrict__ A,    // [NTOK, HID] f16
    const _Float16* __restrict__ Woh,  // [HID, HID] f16 (row = out feature)
    float* __restrict__ out)           // [NTOK, HID] f32
{
    const int lane  = threadIdx.x & 31;
    const int n     = lane & 15;
    const int hi    = lane >> 4;
    const int koffA = hi ? 16 : 0;
    const int koffB = hi ? 16 : 0;

    const int mt = blockIdx.x;
    const int nt = blockIdx.y;

    const _Float16* arow[2] = {
        A + (size_t)(mt * 32 +      n) * HID_,
        A + (size_t)(mt * 32 + 16 + n) * HID_
    };
    const _Float16* W = Woh + (size_t)(nt * 64) * HID_;

    v8f acc[2][4] = {};
    F16x16 aP[2], bP[4], aQ[2], bQ[4];

    auto loadAB = [&](F16x16 (&aa)[2], F16x16 (&bb)[4], int k0) {
#pragma unroll
        for (int mi = 0; mi < 2; ++mi) {
            aa[mi].h[0] = *(const v8h*)(arow[mi] + k0 + koffA);
            aa[mi].h[1] = *(const v8h*)(arow[mi] + k0 + 8 + koffA);
        }
#pragma unroll
        for (int jj = 0; jj < 4; ++jj) {
            const _Float16* brow = W + (size_t)(jj * 16 + n) * HID_ + k0 + koffB;
            bb[jj].h[0] = *(const v8h*)(brow);
            bb[jj].h[1] = *(const v8h*)(brow + 8);
        }
    };
    auto mma8 = [&](F16x16 (&aa)[2], F16x16 (&bb)[4]) {
#pragma unroll
        for (int jj = 0; jj < 4; ++jj) {
            acc[0][jj] = wmma_f16(aa[0].v, bb[jj].v, acc[0][jj]);
            acc[1][jj] = wmma_f16(aa[1].v, bb[jj].v, acc[1][jj]);
        }
    };

    loadAB(aP, bP, 0);
    for (int k0 = 0; k0 < HID_; k0 += 64) {
        loadAB(aQ, bQ, k0 + 32);
        mma8(aP, bP);
        if (k0 + 64 < HID_) loadAB(aP, bP, k0 + 64);
        mma8(aQ, bQ);
    }

#pragma unroll
    for (int mi = 0; mi < 2; ++mi)
#pragma unroll
        for (int jj = 0; jj < 4; ++jj)
#pragma unroll
            for (int r = 0; r < 8; ++r) {
                int m = mi * 16 + r + hi * 8;
                out[(size_t)(mt * 32 + m) * HID_ + nt * 64 + jj * 16 + n] =
                    acc[mi][jj][r];
            }
}

// ---------------- launcher ----------------
extern "C" void kernel_launch(void* const* d_in, const int* in_sizes, int n_in,
                              void* d_out, int out_size, void* d_ws, size_t ws_size,
                              hipStream_t stream) {
    const float* hs  = (const float*)d_in[0];          // [B,S,HID]
    const int*   pid = (const int*)  d_in[1];          // [B,S]
    const float* Wq  = (const float*)d_in[2];          // [NH*HD, HID]
    const float* Wk  = (const float*)d_in[3];          // [NKV*HD, HID]
    const float* Wv  = (const float*)d_in[4];          // [NKV*HD, HID]
    const float* Wo  = (const float*)d_in[5];          // [HID, NH*HD]
    float* out = (float*)d_out;

    // workspace layout (halfs)
    const size_t nX  = (size_t)NTOK * HID_;            // 6,291,456
    const size_t nWq = (size_t)NH_  * HD_ * HID_;      // 2,359,296
    const size_t nWk = (size_t)NKV_ * HD_ * HID_;      //   589,824
    const size_t nWv = nWk;
    const size_t nWo = (size_t)HID_ * NH_ * HD_;       // 2,359,296
    const size_t nQ  = (size_t)B_ * NH_  * S_ * HD_;   // 6,291,456
    const size_t nK  = (size_t)B_ * NKV_ * S_ * HD_;   // 1,572,864
    const size_t nV  = nK;

    _Float16* ws  = (_Float16*)d_ws;
    _Float16* Xh  = ws;              ws += nX;
    _Float16* Wqh = ws;              ws += nWq;
    _Float16* Wkh = ws;              ws += nWk;
    _Float16* Wvh = ws;              ws += nWv;
    _Float16* Woh = ws;              ws += nWo;
    _Float16* Qh  = ws;              ws += nQ;
    _Float16* Kh  = ws;              ws += nK;
    _Float16* Vt  = ws;              ws += nV;
    _Float16* Ao  = ws;              ws += nX;         // attention out, [NTOK, NH*HD]

    // 0) pack fp32 -> f16
    pack_f16_kernel<<<4096, 256, 0, stream>>>(hs, Xh,  (int)nX);
    pack_f16_kernel<<<2048, 256, 0, stream>>>(Wq, Wqh, (int)nWq);
    pack_f16_kernel<<<1024, 256, 0, stream>>>(Wk, Wkh, (int)nWk);
    pack_f16_kernel<<<1024, 256, 0, stream>>>(Wv, Wvh, (int)nWv);
    pack_f16_kernel<<<2048, 256, 0, stream>>>(Wo, Woh, (int)nWo);

    // 1) QKV projection + RoPE (+ V transpose)
    qkv_rope_kernel<<<dim3(NTOK / 32, NH_ + 2 * NKV_), 32, 0, stream>>>(
        Xh, Wqh, Wkh, Wvh, pid, Qh, Kh, Vt);

    // 2) causal flash attention
    attn_kernel<<<dim3(S_ / 64, NH_, B_), 128, 0, stream>>>(Qh, Kh, Vt, Ao);

    // 3) output projection
    oproj_kernel<<<dim3(NTOK / 32, HID_ / 64), 32, 0, stream>>>(Ao, Woh, out);
}
